// BinarizedLinear_54082228191695
// MI455X (gfx1250) — matrix operand order repaired and checked by reading, not compiled, not run on hardware
//
#include <hip/hip_runtime.h>

// ---------------------------------------------------------------------------
// BinarizedLinear on MI455X (gfx1250, wave32):
//   y[t][o] = sum_i x[t][i] * sign(W[o][i])
// GEMM: M=8192, N=4096, K=4096.
//
// Fast path (needs 160MB workspace):
//   1) prepass: X -> (Xhi, Xlo) bf16 split (exact sum ~16 mantissa bits),
//               W -> sign(W) bf16 (exact).  One-time conversion VALU.
//   2) GEMM: pure bf16 tiles, global->LDS via global_load_async_to_lds_b128
//      (ASYNCcnt), double-buffered, v_wmma_f32_16x16x32_bf16 compute.
// Fallback (small ws): fused conversion kernel from round 1.
// ---------------------------------------------------------------------------

typedef __attribute__((ext_vector_type(16))) __bf16 v16bf;
typedef __attribute__((ext_vector_type(8)))  __bf16 v8bf;
typedef __attribute__((ext_vector_type(8)))  float  v8f;

#define TOKENS 8192
#define KDIM   4096
#define NDIM   4096

#define BM 128
#define BN 128
#define BK 32
#define LDT 40      // LDS row stride in bf16 (80B: 16B-aligned chunks, conflict-light)

static __device__ __forceinline__ v16bf pack16(v8bf lo, v8bf hi) {
    v16bf r;
#pragma unroll
    for (int i = 0; i < 8; ++i) { r[i] = lo[i]; r[8 + i] = hi[i]; }
    return r;
}

// Flat address of LDS object: low 32 bits are the LDS byte offset (aperture map).
static __device__ __forceinline__ uint32_t lds_off(const void* p) {
    return (uint32_t)(uintptr_t)p;
}
// gfx1250 async copy: LDS[vdst] = MEM[vaddr], tracked with ASYNCcnt.
static __device__ __forceinline__ void async_b128(uint32_t lds, const void* g) {
    asm volatile("global_load_async_to_lds_b128 %0, %1, off"
                 :: "v"(lds), "v"(g) : "memory");
}
static __device__ __forceinline__ void wait_async0() {
    asm volatile("s_wait_asynccnt 0x0" ::: "memory");
}

// ---------------------------------------------------------------- prepass ---
__global__ __launch_bounds__(256) void prep_x(const float* __restrict__ X,
                                              __bf16* __restrict__ Xhi,
                                              __bf16* __restrict__ Xlo) {
    const size_t i = ((size_t)blockIdx.x * 256 + threadIdx.x) * 4;
    const float4 v = *(const float4*)(X + i);
    const float f[4] = {v.x, v.y, v.z, v.w};
    union { __bf16 h[4]; uint2 u; } H, L;
#pragma unroll
    for (int e = 0; e < 4; ++e) {
        H.h[e] = (__bf16)f[e];
        L.h[e] = (__bf16)(f[e] - (float)H.h[e]);
    }
    *(uint2*)(Xhi + i) = H.u;
    *(uint2*)(Xlo + i) = L.u;
}

__global__ __launch_bounds__(256) void prep_w(const float* __restrict__ W,
                                              __bf16* __restrict__ Ws) {
    const size_t i = ((size_t)blockIdx.x * 256 + threadIdx.x) * 4;
    const float4 v = *(const float4*)(W + i);
    const float f[4] = {v.x, v.y, v.z, v.w};
    union { __bf16 h[4]; uint2 u; } S;
#pragma unroll
    for (int e = 0; e < 4; ++e) {
        const float s = (f[e] > 0.0f) ? 1.0f : ((f[e] < 0.0f) ? -1.0f : 0.0f);
        S.h[e] = (__bf16)s;
    }
    *(uint2*)(Ws + i) = S.u;
}

// ---------------------------------------------------- async-pipelined GEMM ---
__global__ __launch_bounds__(256) void binlin_gemm_async(
    const __bf16* __restrict__ Xhi, const __bf16* __restrict__ Xlo,
    const __bf16* __restrict__ Ws, float* __restrict__ Y)
{
    __shared__ __bf16 sAhi[2][BM * LDT];
    __shared__ __bf16 sAlo[2][BM * LDT];
    __shared__ __bf16 sB  [2][BN * LDT];

    const int tid  = threadIdx.x;
    const int lane = tid & 31;
    const int wid  = tid >> 5;
    const int bm = blockIdx.x * BM;
    const int bn = blockIdx.y * BN;
    const int wm = (wid & 3) * 32;
    const int wn = (wid >> 2) * 64;

    // WMMA fragment addressing (ISA 7.12.2, wave32 bf16 16x16x32)
    const int half = lane >> 4;
    const int fr   = lane & 15;
    const int kA0  = half * 8;
    const int kA1  = 16 + half * 8;
    const int kB   = half * 16;

    // staging: 128 rows x 4 chunks(16B) per array; thread owns 2 chunk-ids
    const int r0 = tid >> 2,          c0 = (tid & 3) * 8;
    const int r1 = (tid + 256) >> 2,  c1 = (tid & 3) * 8;

    v8f acc[2][4];
#pragma unroll
    for (int i = 0; i < 2; ++i)
#pragma unroll
        for (int j = 0; j < 4; ++j)
#pragma unroll
            for (int v = 0; v < 8; ++v) acc[i][j][v] = 0.0f;

    auto stage = [&](int buf, int k0) {
        async_b128(lds_off(&sAhi[buf][r0 * LDT + c0]), Xhi + (size_t)(bm + r0) * KDIM + k0 + c0);
        async_b128(lds_off(&sAhi[buf][r1 * LDT + c1]), Xhi + (size_t)(bm + r1) * KDIM + k0 + c1);
        async_b128(lds_off(&sAlo[buf][r0 * LDT + c0]), Xlo + (size_t)(bm + r0) * KDIM + k0 + c0);
        async_b128(lds_off(&sAlo[buf][r1 * LDT + c1]), Xlo + (size_t)(bm + r1) * KDIM + k0 + c1);
        async_b128(lds_off(&sB [buf][r0 * LDT + c0]),  Ws  + (size_t)(bn + r0) * KDIM + k0 + c0);
        async_b128(lds_off(&sB [buf][r1 * LDT + c1]),  Ws  + (size_t)(bn + r1) * KDIM + k0 + c1);
    };

    stage(0, 0);
    wait_async0();
    __syncthreads();

    const int NT = KDIM / BK;
    for (int kt = 0; kt < NT; ++kt) {
        const int buf = kt & 1;
        if (kt + 1 < NT) stage(buf ^ 1, (kt + 1) * BK);   // prefetch next tile

        v16bf aHi[2], aLo[2], bF[4];
#pragma unroll
        for (int i = 0; i < 2; ++i) {
            const int row = (wm + i * 16 + fr) * LDT;
            aHi[i] = pack16(*(const v8bf*)&sAhi[buf][row + kA0],
                            *(const v8bf*)&sAhi[buf][row + kA1]);
            aLo[i] = pack16(*(const v8bf*)&sAlo[buf][row + kA0],
                            *(const v8bf*)&sAlo[buf][row + kA1]);
        }
#pragma unroll
        for (int j = 0; j < 4; ++j) {
            const int col = (wn + j * 16 + fr) * LDT;
            bF[j] = pack16(*(const v8bf*)&sB[buf][col + kA0],
                           *(const v8bf*)&sB[buf][col + kA1]);
        }
#pragma unroll
        for (int i = 0; i < 2; ++i)
#pragma unroll
            for (int j = 0; j < 4; ++j) {
                acc[i][j] = __builtin_amdgcn_wmma_f32_16x16x32_bf16(
                    false, aHi[i], false, bF[j], (short)0, acc[i][j], false, false);
                acc[i][j] = __builtin_amdgcn_wmma_f32_16x16x32_bf16(
                    false, aLo[i], false, bF[j], (short)0, acc[i][j], false, false);
            }

        wait_async0();      // next tile landed in LDS
        __syncthreads();    // all waves done reading current tile
    }

#pragma unroll
    for (int i = 0; i < 2; ++i)
#pragma unroll
        for (int j = 0; j < 4; ++j) {
            const int m0 = bm + wm + i * 16 + half * 8;
            const int n  = bn + wn + j * 16 + fr;
#pragma unroll
            for (int v = 0; v < 8; ++v)
                __builtin_nontemporal_store(acc[i][j][v], &Y[(size_t)(m0 + v) * NDIM + n]);
        }
}

// ----------------------------------------------- fused fallback (small ws) ---
__global__ __launch_bounds__(256) void binlin_fused(
    const float* __restrict__ X, const float* __restrict__ W, float* __restrict__ Y)
{
    __shared__ __bf16 sAhi[BM * LDT];
    __shared__ __bf16 sAlo[BM * LDT];
    __shared__ __bf16 sB  [BN * LDT];

    const int tid  = threadIdx.x;
    const int lane = tid & 31;
    const int wid  = tid >> 5;
    const int bm = blockIdx.x * BM;
    const int bn = blockIdx.y * BN;
    const int wm = (wid & 3) * 32;
    const int wn = (wid >> 2) * 64;

    v8f acc[2][4];
#pragma unroll
    for (int i = 0; i < 2; ++i)
#pragma unroll
        for (int j = 0; j < 4; ++j)
#pragma unroll
            for (int v = 0; v < 8; ++v) acc[i][j][v] = 0.0f;

    const int gc = tid & 7;
    const int gr = tid >> 3;
    const int half = lane >> 4;
    const int fr   = lane & 15;
    const int kA0  = half * 8;
    const int kA1  = 16 + half * 8;
    const int kB   = half * 16;

    for (int k0 = 0; k0 < KDIM; k0 += BK) {
#pragma unroll
        for (int rr = 0; rr < 4; ++rr) {
            const int r = gr + rr * 32;
            const float4 xa = *(const float4*)(X + (size_t)(bm + r) * KDIM + k0 + gc * 4);
            const float xs[4] = {xa.x, xa.y, xa.z, xa.w};
#pragma unroll
            for (int e = 0; e < 4; ++e) {
                const __bf16 h = (__bf16)xs[e];
                sAhi[r * LDT + gc * 4 + e] = h;
                sAlo[r * LDT + gc * 4 + e] = (__bf16)(xs[e] - (float)h);
            }
            const float4 wa = *(const float4*)(W + (size_t)(bn + r) * KDIM + k0 + gc * 4);
            const float ws[4] = {wa.x, wa.y, wa.z, wa.w};
#pragma unroll
            for (int e = 0; e < 4; ++e) {
                const float s = (ws[e] > 0.0f) ? 1.0f : ((ws[e] < 0.0f) ? -1.0f : 0.0f);
                sB[r * LDT + gc * 4 + e] = (__bf16)s;
            }
        }
        __syncthreads();

        v16bf aHi[2], aLo[2], bF[4];
#pragma unroll
        for (int i = 0; i < 2; ++i) {
            const int row = (wm + i * 16 + fr) * LDT;
            aHi[i] = pack16(*(const v8bf*)&sAhi[row + kA0], *(const v8bf*)&sAhi[row + kA1]);
            aLo[i] = pack16(*(const v8bf*)&sAlo[row + kA0], *(const v8bf*)&sAlo[row + kA1]);
        }
#pragma unroll
        for (int j = 0; j < 4; ++j) {
            const int col = (wn + j * 16 + fr) * LDT;
            bF[j] = pack16(*(const v8bf*)&sB[col + kA0], *(const v8bf*)&sB[col + kA1]);
        }
#pragma unroll
        for (int i = 0; i < 2; ++i)
#pragma unroll
            for (int j = 0; j < 4; ++j) {
                acc[i][j] = __builtin_amdgcn_wmma_f32_16x16x32_bf16(
                    false, aHi[i], false, bF[j], (short)0, acc[i][j], false, false);
                acc[i][j] = __builtin_amdgcn_wmma_f32_16x16x32_bf16(
                    false, aLo[i], false, bF[j], (short)0, acc[i][j], false, false);
            }
        __syncthreads();
    }

#pragma unroll
    for (int i = 0; i < 2; ++i)
#pragma unroll
        for (int j = 0; j < 4; ++j) {
            const int m0 = bm + wm + i * 16 + half * 8;
            const int n  = bn + wn + j * 16 + fr;
#pragma unroll
            for (int v = 0; v < 8; ++v)
                Y[(size_t)(m0 + v) * NDIM + n] = acc[i][j][v];
        }
}

// ------------------------------------------------------------------ launch ---
extern "C" void kernel_launch(void* const* d_in, const int* in_sizes, int n_in,
                              void* d_out, int out_size, void* d_ws, size_t ws_size,
                              hipStream_t stream) {
    (void)in_sizes; (void)n_in; (void)out_size;
    const float* X = (const float*)d_in[0];
    const float* W = (const float*)d_in[1];
    float* Y = (float*)d_out;

    const size_t nX = (size_t)TOKENS * KDIM;   // 33.5M
    const size_t nW = (size_t)NDIM * KDIM;     // 16.7M
    const size_t ws_need = (2 * nX + nW) * sizeof(__bf16);  // 160 MB

    dim3 grid(TOKENS / BM, NDIM / BN);
    if (ws_size >= ws_need) {
        __bf16* Xhi = (__bf16*)d_ws;
        __bf16* Xlo = Xhi + nX;
        __bf16* Ws  = Xlo + nX;
        prep_x<<<dim3((unsigned)(nX / 4 / 256)), dim3(256), 0, stream>>>(X, Xhi, Xlo);
        prep_w<<<dim3((unsigned)(nW / 4 / 256)), dim3(256), 0, stream>>>(W, Ws);
        binlin_gemm_async<<<grid, dim3(256), 0, stream>>>(Xhi, Xlo, Ws, Y);
    } else {
        binlin_fused<<<grid, dim3(256), 0, stream>>>(X, W, Y);
    }
}
